// SeededAttention_35407710389010
// MI455X (gfx1250) — compile-verified, matrix operand order checked by reading, not run on hardware
//
#include <hip/hip_runtime.h>
#include <hip/hip_bf16.h>

// ---------------- problem constants ----------------
#define BQ   8
#define TQ   512
#define CQ   2048
#define HH   16
#define HDQ  128
#define RANKQ 8
#define RDQ  16
#define N_QKV (3 * CQ)          // 6144
#define M_ROWS (BQ * TQ)        // 4096
#define ADAPTER_SCALE (1.0f / 8.0f)
#define QSCALE 0.08838834764831845f   // 1/sqrt(128)

typedef __attribute__((ext_vector_type(16))) __bf16 v16bf;
typedef __attribute__((ext_vector_type(8)))  float  v8f;

union Frag {
    v16bf v;
    uint4 q[2];
    unsigned u[8];
    unsigned short e[16];
};

__device__ __forceinline__ unsigned short f2bf(float f) {
    unsigned u = __float_as_uint(f);
    return (unsigned short)((u + 0x8000u) >> 16);
}

__device__ __forceinline__ v8f zero8() {
    v8f v;
#pragma unroll
    for (int i = 0; i < 8; ++i) v[i] = 0.0f;
    return v;
}

__device__ __forceinline__ v16bf load_frag_lds(const unsigned short* p) {
    Frag f;
    f.q[0] = *(const uint4*)(p);
    f.q[1] = *(const uint4*)(p + 16);
    return f.v;
}

// low 32 bits of a generic LDS pointer = wave-relative LDS byte offset
__device__ __forceinline__ unsigned lds_off(const void* p) {
    return (unsigned)(unsigned long long)(uintptr_t)p;
}

// CDNA5 async global->LDS copy, 16 bytes per lane, tracked by ASYNCcnt
__device__ __forceinline__ void async_copy16(unsigned lds_dst, const void* gsrc) {
    asm volatile("global_load_async_to_lds_b128 %0, %1, off"
                 :: "v"(lds_dst), "v"((unsigned long long)(uintptr_t)gsrc)
                 : "memory");
}

__device__ __forceinline__ void wait_async0() {
    asm volatile("s_wait_asynccnt 0" ::: "memory");
}

// pack the top halves of two f32 into one dword: {b[31:16], a[31:16]}
__device__ __forceinline__ unsigned pack_hi16(unsigned ua, unsigned ub) {
    // v_perm_b32: src0 supplies bytes 7:4, src1 supplies bytes 3:0
    return __builtin_amdgcn_perm(ub, ua, 0x07060302u);
}

// build bf16 hi/lo fragments from 16 f32 in LDS (runs at +0..7 and +16..23)
// packed-dword construction: 2x v_perm_b32 + residual math per f32 pair
__device__ __forceinline__ void build_frag(const float* b0, v16bf& hi, v16bf& lo) {
    float f[16];
    *(float4*)(f + 0)  = *(const float4*)(b0);
    *(float4*)(f + 4)  = *(const float4*)(b0 + 4);
    *(float4*)(f + 8)  = *(const float4*)(b0 + 16);
    *(float4*)(f + 12) = *(const float4*)(b0 + 20);
    Frag H, L;
#pragma unroll
    for (int i = 0; i < 8; ++i) {
        float a = f[2 * i], b = f[2 * i + 1];
        unsigned ua = __float_as_uint(a);
        unsigned ub = __float_as_uint(b);
        H.u[i] = pack_hi16(ua, ub);
        float fla = a - __uint_as_float(ua & 0xffff0000u);
        float flb = b - __uint_as_float(ub & 0xffff0000u);
        L.u[i] = pack_hi16(__float_as_uint(fla), __float_as_uint(flb));
    }
    hi = H.v;
    lo = L.v;
}

// ---------------- kernel 1: LoRA h = x @ A0^T * (da*db) ----------------
__global__ void lora_h_kernel(const float* __restrict__ X, const float* __restrict__ A0,
                              const float* __restrict__ da, const float* __restrict__ db,
                              float* __restrict__ h, int M, int K) {
    int idx = blockIdx.x * blockDim.x + threadIdx.x;   // m*RANK + r
    if (idx >= M * RANKQ) return;
    int m = idx / RANKQ;
    int r = idx % RANKQ;
    const float* xr = X + (size_t)m * K;
    const float* ar = A0 + (size_t)r * K;
    float s = 0.0f;
    for (int k = 0; k < K; k += 4) {
        float4 xv = *(const float4*)(xr + k);
        float4 av = *(const float4*)(ar + k);
        s += xv.x * av.x + xv.y * av.y + xv.z * av.z + xv.w * av.w;
    }
    h[idx] = s * da[r] * db[r];
}

// ---------------- kernel 2: GEMM D = X @ W^T (fp32 in/out, 3-term bf16 WMMA) ----------------
// f32 tiles staged via async global->LDS copies; bf16 hi/lo split at fragment build.
#define BM 128
#define BN 128
#define BK 32
#define LDF 36   // f32 LDS row stride (floats): 144B, 16B aligned

__global__ __launch_bounds__(256) void gemm_bf16x3_kernel(const float* __restrict__ X,
                                                          const float* __restrict__ W,
                                                          float* __restrict__ D,
                                                          int M, int N, int K) {
    __shared__ float sX[BM * LDF];
    __shared__ float sW[BN * LDF];

    const int tid  = threadIdx.x;
    const int m0   = blockIdx.y * BM;
    const int n0   = blockIdx.x * BN;
    const int wid  = tid >> 5;
    const int lane = tid & 31;
    const int wm   = (wid & 3) * 32;   // wave row offset in block (4 waves along M)
    const int wn   = (wid >> 2) * 64;  // wave col offset in block (2 waves along N)
    const int lr   = lane & 15;
    const int lh   = lane >> 4;

    v8f acc[2][4];
#pragma unroll
    for (int a = 0; a < 2; ++a)
#pragma unroll
        for (int b = 0; b < 4; ++b) acc[a][b] = zero8();

    for (int kb = 0; kb < K; kb += BK) {
        // async-stage X[128][32] and W[128][32] f32 tiles into LDS
#pragma unroll
        for (int i = 0; i < 4; ++i) {
            int idx = tid + i * 256;            // 1024 x 16B chunks per matrix
            int row = idx >> 3;                 // 0..127
            int c4  = (idx & 7) * 4;            // 0,4,..,28
            async_copy16(lds_off(&sX[row * LDF + c4]),
                         X + (size_t)(m0 + row) * K + kb + c4);
            async_copy16(lds_off(&sW[row * LDF + c4]),
                         W + (size_t)(n0 + row) * K + kb + c4);
        }
        // prefetch next K tile into L2 while we compute
        if (kb + BK < K) {
            __builtin_prefetch(X + (size_t)(m0 + (tid >> 1)) * K + kb + BK, 0, 0);
            __builtin_prefetch(W + (size_t)(n0 + (tid >> 1)) * K + kb + BK, 0, 0);
        }
        wait_async0();
        __syncthreads();

        // A fragments (hi/lo) for this wave's two 16-row tiles
        v16bf Ah[2], Al[2];
#pragma unroll
        for (int tm = 0; tm < 2; ++tm)
            build_frag(&sX[(wm + tm * 16 + lr) * LDF + lh * 8], Ah[tm], Al[tm]);

#pragma unroll
        for (int tn = 0; tn < 4; ++tn) {
            v16bf Bh, Bl;
            build_frag(&sW[(wn + tn * 16 + lr) * LDF + lh * 8], Bh, Bl);
#pragma unroll
            for (int tm = 0; tm < 2; ++tm) {
                acc[tm][tn] = __builtin_amdgcn_wmma_f32_16x16x32_bf16(
                    false, Ah[tm], false, Bh, (short)0, acc[tm][tn], false, false);
                acc[tm][tn] = __builtin_amdgcn_wmma_f32_16x16x32_bf16(
                    false, Ah[tm], false, Bl, (short)0, acc[tm][tn], false, false);
                acc[tm][tn] = __builtin_amdgcn_wmma_f32_16x16x32_bf16(
                    false, Al[tm], false, Bh, (short)0, acc[tm][tn], false, false);
            }
        }
        __syncthreads();
    }

    // epilogue: C-layout element (vgpr i, lane L) -> row = i + 8*(L>=16), col = L%16
#pragma unroll
    for (int tm = 0; tm < 2; ++tm)
#pragma unroll
        for (int tn = 0; tn < 4; ++tn) {
#pragma unroll
            for (int i = 0; i < 8; ++i) {
                int m = m0 + wm + tm * 16 + i + lh * 8;
                int n = n0 + wn + tn * 16 + lr;
                D[(size_t)m * N + n] = acc[tm][tn][i];
            }
        }
}

// ---------------- kernel 3: rank-8 LoRA add D += scale * h @ B0^T ----------------
__global__ void lora_add_kernel(float* __restrict__ D, const float* __restrict__ h,
                                const float* __restrict__ B0, int M, int N, float scale) {
    size_t total = (size_t)M * N;
    for (size_t idx = (size_t)blockIdx.x * blockDim.x + threadIdx.x; idx < total;
         idx += (size_t)gridDim.x * blockDim.x) {
        int m = (int)(idx / N);
        int n = (int)(idx % N);
        const float* hr = h + (size_t)m * RANKQ;
        const float* br = B0 + (size_t)n * RANKQ;
        float s = 0.0f;
#pragma unroll
        for (int r = 0; r < RANKQ; ++r) s += hr[r] * br[r];
        D[idx] += s * scale;
    }
}

// ---------------- kernel 4: finalize qkv -> q(rope,scaled) / k(rope) / v^T, bf16 ----------------
__global__ void finalize_kernel(const float* __restrict__ qkv,
                                const float* __restrict__ cosb, const float* __restrict__ sinb,
                                unsigned short* __restrict__ qb, unsigned short* __restrict__ kb_,
                                unsigned short* __restrict__ vtb) {
    int row = blockIdx.x;           // b*T + t
    int t = row % TQ;
    int b = row / TQ;
    const float* base = qkv + (size_t)row * N_QKV;
    const int half = RDQ / 2;
    for (int c = threadIdx.x; c < CQ; c += blockDim.x) {
        int hh = c / HDQ;
        int d  = c % HDQ;
        size_t bh = (size_t)b * HH + hh;
        float qv = base[c];
        float kv = base[CQ + c];
        float vv = base[2 * CQ + c];
        if (d < RDQ) {
            if (d < half) {
                float cs = cosb[t * half + d];
                float sn = sinb[t * half + d];
                float q2 = base[c + half];
                float k2 = base[CQ + c + half];
                qv = qv * cs - q2 * sn;
                kv = kv * cs - k2 * sn;
            } else {
                float cs = cosb[t * half + d - half];
                float sn = sinb[t * half + d - half];
                float q1 = base[c - half];
                float k1 = base[CQ + c - half];
                qv = qv * cs + q1 * sn;
                kv = kv * cs + k1 * sn;
            }
        }
        qb[(bh * TQ + t) * HDQ + d] = f2bf(qv * QSCALE);
        kb_[(bh * TQ + t) * HDQ + d] = f2bf(kv);
        vtb[(bh * HDQ + d) * TQ + t] = f2bf(vv);
    }
}

// ---------------- kernel 5: causal flash attention (bf16 WMMA, async K/V staging) ----------------
#define KLD 136   // k LDS row stride (ushorts): 272B, 16B aligned
#define VLD 72    // v / P LDS row stride (ushorts): 144B, 16B aligned

__global__ __launch_bounds__(128) void attn_kernel(const unsigned short* __restrict__ qb,
                                                   const unsigned short* __restrict__ kbuf,
                                                   const unsigned short* __restrict__ vtb,
                                                   float* __restrict__ Y) {
    __shared__ unsigned short kS[64 * KLD];
    __shared__ unsigned short vS[128 * VLD];
    __shared__ unsigned short pS[4 * 16 * VLD];

    const int bh   = blockIdx.y;       // 0..127
    const int qt   = blockIdx.x;       // 0..7  (64-query tiles)
    const int tid  = threadIdx.x;
    const int wid  = tid >> 5;
    const int lane = tid & 31;
    const int lr   = lane & 15;
    const int lh   = lane >> 4;
    const int q0   = qt * 64 + wid * 16;     // wave's first query row

    const unsigned short* qbase = qb  + (size_t)bh * TQ * HDQ;
    const unsigned short* kbase = kbuf + (size_t)bh * TQ * HDQ;
    const unsigned short* vbase = vtb + (size_t)bh * HDQ * TQ;

    // load Q A-fragments for this wave (rows q0..q0+15, K = HD in 4 chunks of 32)
    v16bf Qf[4];
#pragma unroll
    for (int kc = 0; kc < 4; ++kc) {
        const unsigned short* p = qbase + (size_t)(q0 + lr) * HDQ + kc * 32 + lh * 8;
        Frag f;
        f.q[0] = *(const uint4*)p;
        f.q[1] = *(const uint4*)(p + 16);
        Qf[kc] = f.v;
    }

    v8f O[8];
#pragma unroll
    for (int i = 0; i < 8; ++i) O[i] = zero8();
    float mrun[8], lrun[8];
#pragma unroll
    for (int i = 0; i < 8; ++i) { mrun[i] = -1e30f; lrun[i] = 0.0f; }

    for (int kb2 = 0; kb2 <= qt; ++kb2) {
        const int key0 = kb2 * 64;
        // async-stage K block [64 keys][128 d] and V block [128 d][64 keys] in LDS
#pragma unroll
        for (int i = 0; i < 8; ++i) {
            int idx = tid + i * 128;
            int row = idx >> 4;             // 0..63
            int c8  = (idx & 15) * 8;       // 0..120
            async_copy16(lds_off(&kS[row * KLD + c8]),
                         kbase + (size_t)(key0 + row) * HDQ + c8);
        }
#pragma unroll
        for (int i = 0; i < 8; ++i) {
            int idx = tid + i * 128;
            int row = idx >> 3;             // 0..127
            int c8  = (idx & 7) * 8;        // 0..56
            async_copy16(lds_off(&vS[row * VLD + c8]),
                         vbase + (size_t)row * TQ + key0 + c8);
        }
        wait_async0();
        __syncthreads();

        // scores S[16q x 64k] = Q @ K^T
        v8f S[4];
#pragma unroll
        for (int tn = 0; tn < 4; ++tn) {
            v8f s = zero8();
#pragma unroll
            for (int kc = 0; kc < 4; ++kc) {
                v16bf Bf = load_frag_lds(&kS[(tn * 16 + lr) * KLD + kc * 32 + lh * 8]);
                s = __builtin_amdgcn_wmma_f32_16x16x32_bf16(
                    false, Qf[kc], false, Bf, (short)0, s, false, false);
            }
            S[tn] = s;
        }

        // causal mask (only the diagonal block needs it)
        if (kb2 == qt) {
#pragma unroll
            for (int tn = 0; tn < 4; ++tn)
#pragma unroll
                for (int i = 0; i < 8; ++i) {
                    int key  = key0 + tn * 16 + lr;
                    int qrow = q0 + i + lh * 8;
                    if (key > qrow) S[tn][i] = -1e30f;
                }
        }

        // online softmax: rows are striped across 16 lanes in C-layout
        float mnew[8], scl[8];
#pragma unroll
        for (int i = 0; i < 8; ++i) {
            float rm = fmaxf(fmaxf(S[0][i], S[1][i]), fmaxf(S[2][i], S[3][i]));
#pragma unroll
            for (int off = 8; off >= 1; off >>= 1)
                rm = fmaxf(rm, __shfl_xor(rm, off, 16));
            mnew[i] = fmaxf(mrun[i], rm);
            scl[i]  = __expf(mrun[i] - mnew[i]);
            mrun[i] = mnew[i];
        }
        float rs[8];
#pragma unroll
        for (int i = 0; i < 8; ++i) rs[i] = 0.0f;
#pragma unroll
        for (int tn = 0; tn < 4; ++tn)
#pragma unroll
            for (int i = 0; i < 8; ++i) {
                float p = __expf(S[tn][i] - mnew[i]);
                S[tn][i] = p;
                rs[i] += p;
            }
#pragma unroll
        for (int i = 0; i < 8; ++i) {
            float r = rs[i];
#pragma unroll
            for (int off = 8; off >= 1; off >>= 1)
                r += __shfl_xor(r, off, 16);
            lrun[i] = lrun[i] * scl[i] + r;
        }
#pragma unroll
        for (int ot = 0; ot < 8; ++ot)
#pragma unroll
            for (int i = 0; i < 8; ++i) O[ot][i] *= scl[i];

        // transpose P (C-layout) -> A-layout through this wave's LDS region
        unsigned short* pw = &pS[wid * 16 * VLD];
#pragma unroll
        for (int tn = 0; tn < 4; ++tn)
#pragma unroll
            for (int i = 0; i < 8; ++i)
                pw[(i + lh * 8) * VLD + tn * 16 + lr] = f2bf(S[tn][i]);

        // O += P @ V
#pragma unroll
        for (int kc = 0; kc < 2; ++kc) {
            v16bf Pf = load_frag_lds(&pS[wid * 16 * VLD + lr * VLD + kc * 32 + lh * 8]);
#pragma unroll
            for (int ot = 0; ot < 8; ++ot) {
                v16bf Vf = load_frag_lds(&vS[(ot * 16 + lr) * VLD + kc * 32 + lh * 8]);
                O[ot] = __builtin_amdgcn_wmma_f32_16x16x32_bf16(
                    false, Pf, false, Vf, (short)0, O[ot], false, false);
            }
        }
        __syncthreads();
    }

    // epilogue: y[b, t, h*HD + d] = O / l
    const int b  = bh / HH;
    const int hh = bh % HH;
#pragma unroll
    for (int i = 0; i < 8; ++i) {
        float inv = 1.0f / lrun[i];
        int trow = q0 + i + lh * 8;
        float* yrow = Y + ((size_t)b * TQ + trow) * CQ + hh * HDQ;
#pragma unroll
        for (int ot = 0; ot < 8; ++ot)
            yrow[ot * 16 + lr] = O[ot][i] * inv;
    }
}

// ---------------- host launch ----------------
extern "C" void kernel_launch(void* const* d_in, const int* in_sizes, int n_in,
                              void* d_out, int out_size, void* d_ws, size_t ws_size,
                              hipStream_t stream) {
    const float* x    = (const float*)d_in[0];
    const float* Wqkv = (const float*)d_in[1];
    const float* A0q  = (const float*)d_in[2];
    const float* B0q  = (const float*)d_in[3];
    const float* daq  = (const float*)d_in[4];
    const float* dbq  = (const float*)d_in[5];
    const float* Wout = (const float*)d_in[6];
    const float* A0o  = (const float*)d_in[7];
    const float* B0o  = (const float*)d_in[8];
    const float* dao  = (const float*)d_in[9];
    const float* dbo  = (const float*)d_in[10];
    const float* cosb = (const float*)d_in[11];
    const float* sinb = (const float*)d_in[12];
    float* out = (float*)d_out;

    char* ws = (char*)d_ws;
    size_t off = 0;
    float* qkv = (float*)(ws + off);              // [4096, 6144] f32; reused as Y later
    off += (size_t)M_ROWS * N_QKV * sizeof(float);
    unsigned short* qbf = (unsigned short*)(ws + off); off += (size_t)BQ * HH * TQ * HDQ * 2;
    unsigned short* kbf = (unsigned short*)(ws + off); off += (size_t)BQ * HH * TQ * HDQ * 2;
    unsigned short* vtb = (unsigned short*)(ws + off); off += (size_t)BQ * HH * HDQ * TQ * 2;
    float* h_qkv = (float*)(ws + off); off += (size_t)M_ROWS * RANKQ * sizeof(float);
    float* h_out = (float*)(ws + off); off += (size_t)M_ROWS * RANKQ * sizeof(float);
    float* Y = qkv;   // alias: qkv buffer is dead after finalize

    // 1) LoRA hidden for QKV
    lora_h_kernel<<<(M_ROWS * RANKQ + 255) / 256, 256, 0, stream>>>(
        x, A0q, daq, dbq, h_qkv, M_ROWS, CQ);

    // 2) QKV base GEMM: qkv = x @ Wqkv^T
    gemm_bf16x3_kernel<<<dim3(N_QKV / BN, M_ROWS / BM), 256, 0, stream>>>(
        x, Wqkv, qkv, M_ROWS, N_QKV, CQ);

    // 3) qkv += (h_qkv @ B0q^T) / RANK
    lora_add_kernel<<<2048, 256, 0, stream>>>(qkv, h_qkv, B0q, M_ROWS, N_QKV, ADAPTER_SCALE);

    // 4) split + RoPE + transpose-V, cast to bf16
    finalize_kernel<<<M_ROWS, 256, 0, stream>>>(qkv, cosb, sinb, qbf, kbf, vtb);

    // 5) causal flash attention -> Y (f32, [4096, 2048]); overwrites qkv region
    attn_kernel<<<dim3(TQ / 64, BQ * HH), 128, 0, stream>>>(qbf, kbf, vtb, Y);

    // 6) LoRA hidden for output projection
    lora_h_kernel<<<(M_ROWS * RANKQ + 255) / 256, 256, 0, stream>>>(
        Y, A0o, dao, dbo, h_out, M_ROWS, CQ);

    // 7) out = Y @ Wout^T
    gemm_bf16x3_kernel<<<dim3(CQ / BN, M_ROWS / BM), 256, 0, stream>>>(
        Y, Wout, out, M_ROWS, CQ, CQ);

    // 8) out += (h_out @ B0o^T) / RANK
    lora_add_kernel<<<1024, 256, 0, stream>>>(out, h_out, B0o, M_ROWS, CQ, ADAPTER_SCALE);
}
